// MultiHeadAttention_22703197126776
// MI455X (gfx1250) — matrix operand order repaired and checked by reading, not compiled, not run on hardware
//
#include <hip/hip_runtime.h>
#include <hip/hip_bf16.h>

// ---- problem constants -------------------------------------------------
#define BB 8
#define NN 1024
#define DD 512
#define HH 16
#define DH 64
#define HD (HH * DH)   // 1024
#define MROWS (BB * NN) // 8192

typedef __attribute__((ext_vector_type(16))) _Float16 v16h;
typedef __attribute__((ext_vector_type(8)))  _Float16 v8h;
typedef __attribute__((ext_vector_type(8)))  float    v8f;

#define DEVINL __device__ __forceinline__

// ---- WMMA wrapper: D = A(16x32 f16) * B(32x16 f16) + C(16x16 f32) ------
DEVINL v8f wmma_f16(v16h a, v16h b, v8f c) {
  return __builtin_amdgcn_wmma_f32_16x16x32_f16(
      /*neg_a=*/false, a, /*neg_b=*/false, b,
      /*c_mod=*/(short)0, c, /*reuse_a=*/false, /*reuse_b=*/false);
}

// ---- A fragment (16x32, f16 source, row-major, stride ld elements) -----
// lanes 0-15: row=lane,   K = {0..7, 16..23}
// lanes16-31: row=lane-16,K = {8..15, 24..31}
DEVINL v16h load_a16(const _Float16* base, int ld) {
  const int lane = threadIdx.x & 31;
  const int r  = lane & 15;
  const int kb = (lane & 16) ? 8 : 0;
  const _Float16* p = base + (size_t)r * ld + kb;
  v8h lo = *(const v8h*)(p);
  v8h hi = *(const v8h*)(p + 16);
  v16h a;
#pragma unroll
  for (int i = 0; i < 8; ++i) { a[i] = lo[i]; a[8 + i] = hi[i]; }
  return a;
}

// ---- A fragment from fp32 source (convert on load) ---------------------
DEVINL v16h load_a32(const float* base, int ld) {
  const int lane = threadIdx.x & 31;
  const int r  = lane & 15;
  const int kb = (lane & 16) ? 8 : 0;
  const float* p = base + (size_t)r * ld + kb;
  v16h a;
#pragma unroll
  for (int i = 0; i < 8; ++i) {
    a[i]     = (_Float16)p[i];
    a[8 + i] = (_Float16)p[16 + i];
  }
  return a;
}

// ---- B fragment (32x16): source bt[c*ld + k] == B[k][c] ----------------
// lanes 0-15: col=lane,    K = {0..7, 16..23}
// lanes16-31: col=lane-16, K = {8..15, 24..31}
DEVINL v16h load_b16(const _Float16* bt, int ld) {
  const int lane = threadIdx.x & 31;
  const int c  = lane & 15;
  const int kb = (lane & 16) ? 8 : 0;
  const _Float16* p = bt + (size_t)c * ld + kb;
  v8h lo = *(const v8h*)(p);
  v8h hi = *(const v8h*)(p + 16);
  v16h b;
#pragma unroll
  for (int i = 0; i < 8; ++i) { b[i] = lo[i]; b[8 + i] = hi[i]; }
  return b;
}

// ---- half-wave (16 lane) reductions; rows of a C tile live per-half ----
DEVINL float half16_max(float v) {
#pragma unroll
  for (int m = 1; m < 16; m <<= 1) v = fmaxf(v, __shfl_xor(v, m, 32));
  return v;
}
DEVINL float half16_sum(float v) {
#pragma unroll
  for (int m = 1; m < 16; m <<= 1) v += __shfl_xor(v, m, 32);
  return v;
}

// ========================================================================
// Kernel 0: transpose fp32 (rows x cols) -> f16 dst[c*rows + r]
// ========================================================================
__global__ __launch_bounds__(256) void k_transpose_w(const float* __restrict__ src,
                                                     _Float16* __restrict__ dst,
                                                     int rows, int cols) {
  int i = blockIdx.x * blockDim.x + threadIdx.x;
  int total = rows * cols;
  if (i >= total) return;
  int r = i / cols, c = i % cols;
  dst[(size_t)c * rows + r] = (_Float16)src[i];
}

// ========================================================================
// Kernel 1: QKV projection.  X(8192x512 f32) @ W(512x1024) + b
//   mat 0 -> Qbf [m][col] f16     (row-major, 1024 stride)
//   mat 1 -> Kbf [m][col] f16
//   mat 2 -> Vt  [b][h][dh][n] f16 (transposed scatter)
// one wave per 16x16 tile; 512*64*3 = 98304 waves
// ========================================================================
__global__ __launch_bounds__(256) void k_qkv(const float* __restrict__ X,
                                             const _Float16* __restrict__ WqT,
                                             const _Float16* __restrict__ WkT,
                                             const _Float16* __restrict__ WvT,
                                             const float* __restrict__ bq,
                                             const float* __restrict__ bk,
                                             const float* __restrict__ bv,
                                             _Float16* __restrict__ Qbf,
                                             _Float16* __restrict__ Kbf,
                                             _Float16* __restrict__ Vt) {
  const int wid  = blockIdx.x * (blockDim.x >> 5) + (threadIdx.x >> 5);
  const int ct   = wid & 63;        // column tile (of 64)
  const int t    = wid >> 6;
  const int mat  = t % 3;
  const int mt   = t / 3;           // row tile (of 512)
  const int lane = threadIdx.x & 31;
  const int lc   = lane & 15;
  const int hf   = lane >> 4;

  const _Float16* wt   = (mat == 0) ? WqT : ((mat == 1) ? WkT : WvT);
  const float*    bias = (mat == 0) ? bq  : ((mat == 1) ? bk  : bv);

  v8f acc = {};
  const float*    arow = X  + (size_t)mt * 16 * DD;
  const _Float16* brow = wt + (size_t)ct * 16 * DD;
#pragma unroll 4
  for (int k0 = 0; k0 < DD; k0 += 32) {
    v16h a = load_a32(arow + k0, DD);
    v16h b = load_b16(brow + k0, DD);
    acc = wmma_f16(a, b, acc);
  }

  const int col = ct * 16 + lc;
  const float bval = bias[col];

  if (mat < 2) {
    _Float16* dst = (mat == 0) ? Qbf : Kbf;
#pragma unroll
    for (int g = 0; g < 8; ++g) {
      int m = mt * 16 + g + 8 * hf;
      dst[(size_t)m * HD + col] = (_Float16)(acc[g] + bval);
    }
  } else {
    const int h  = col >> 6;
    const int dh = col & 63;
#pragma unroll
    for (int g = 0; g < 8; ++g) {
      int m = mt * 16 + g + 8 * hf;
      int b_ = m >> 10;
      int n  = m & 1023;
      Vt[((size_t)(b_ * HH + h) * DH + dh) * NN + n] = (_Float16)(acc[g] + bval);
    }
  }
}

// ========================================================================
// Kernel 2: flash attention. One wave per (b, h, 16-query tile) = 8192 waves
// ========================================================================
__global__ __launch_bounds__(256) void k_attn(const _Float16* __restrict__ Qbf,
                                              const _Float16* __restrict__ Kbf,
                                              const _Float16* __restrict__ Vt,
                                              const unsigned char* __restrict__ mask,
                                              _Float16* __restrict__ Ctx) {
  __shared__ float sm[8][16 * 32];
  const int warp = threadIdx.x >> 5;
  const int wid  = blockIdx.x * 8 + warp;
  const int qt = wid & 63;
  const int h  = (wid >> 6) & 15;
  const int b  = wid >> 10;
  const int q0 = qt * 16;
  const int lane = threadIdx.x & 31;
  const int lc = lane & 15;
  const int hf = lane >> 4;
  float* my = sm[warp];

  const _Float16* Qbase = Qbf + ((size_t)(b * NN + q0)) * HD + h * DH;
  v16h aq0 = load_a16(Qbase, HD);        // dh 0..31
  v16h aq1 = load_a16(Qbase + 32, HD);   // dh 32..63

  float mrun[8], lrun[8];
  v8f acc[4] = {};
#pragma unroll
  for (int g = 0; g < 8; ++g) { mrun[g] = -1.0e30f; lrun[g] = 0.0f; }

  const unsigned char* mbase = mask + ((size_t)b * NN + q0) * NN;

  for (int j = 0; j < NN; j += 32) {
    // ---- scores: two 16x16 tiles (keys j..j+15, j+16..j+31) ----
    const _Float16* Kb0 = Kbf + ((size_t)(b * NN + j)) * HD + h * DH;
    const _Float16* Kb1 = Kb0 + (size_t)16 * HD;
    v8f s0 = {}, s1 = {};
    s0 = wmma_f16(aq0, load_b16(Kb0,      HD), s0);
    s0 = wmma_f16(aq1, load_b16(Kb0 + 32, HD), s0);
    s1 = wmma_f16(aq0, load_b16(Kb1,      HD), s1);
    s1 = wmma_f16(aq1, load_b16(Kb1 + 32, HD), s1);

    // ---- scale + mask ----
#pragma unroll
    for (int g = 0; g < 8; ++g) {
      const int q = g + 8 * hf;
      const unsigned char* mp = mbase + (size_t)q * NN + j + lc;
      float v0 = s0[g] * 0.125f;   // 1/sqrt(64)
      float v1 = s1[g] * 0.125f;
      if (mp[0])  v0 = -1.0e30f;
      if (mp[16]) v1 = -1.0e30f;
      s0[g] = v0; s1[g] = v1;
    }

    // ---- online softmax (rows live in 16-lane halves) ----
#pragma unroll
    for (int g = 0; g < 8; ++g) {
      float mx   = half16_max(fmaxf(s0[g], s1[g]));
      float mnew = fmaxf(mrun[g], mx);
      float sc   = __expf(mrun[g] - mnew);
      mrun[g] = mnew;
      float p0 = __expf(s0[g] - mnew);
      float p1 = __expf(s1[g] - mnew);
      s0[g] = p0; s1[g] = p1;
      lrun[g] = lrun[g] * sc + half16_sum(p0 + p1);
#pragma unroll
      for (int t = 0; t < 4; ++t) acc[t][g] *= sc;
    }

    // ---- stage P (C layout f32) -> LDS -> A layout f16 ----
#pragma unroll
    for (int g = 0; g < 8; ++g) {
      const int row = g + 8 * hf;
      my[row * 32 + lc]      = s0[g];
      my[row * 32 + 16 + lc] = s1[g];
    }
    asm volatile("s_wait_dscnt 0" ::: "memory");
    v16h pa;
    {
      const int kb = hf ? 8 : 0;
#pragma unroll
      for (int i = 0; i < 8; ++i) {
        pa[i]     = (_Float16)my[lc * 32 + kb + i];
        pa[8 + i] = (_Float16)my[lc * 32 + kb + 16 + i];
      }
    }
    asm volatile("" ::: "memory");

    // ---- ctx += P(16x32) @ V(32keys x 64dh) ----
    const _Float16* Vb = Vt + ((size_t)(b * HH + h) * DH) * NN + j;
#pragma unroll
    for (int t = 0; t < 4; ++t)
      acc[t] = wmma_f16(pa, load_b16(Vb + (size_t)t * 16 * NN, NN), acc[t]);
  }

  // ---- epilogue: normalize and store ctx as f16 [b,n,h*64+dh] ----
  _Float16* crow = Ctx + ((size_t)(b * NN + q0)) * HD + h * DH;
#pragma unroll
  for (int g = 0; g < 8; ++g) {
    const float inv = 1.0f / lrun[g];
    const int q = g + 8 * hf;
#pragma unroll
    for (int t = 0; t < 4; ++t)
      crow[(size_t)q * HD + t * 16 + lc] = (_Float16)(acc[t][g] * inv);
  }
}

// ========================================================================
// Kernel 3: output projection. ctx(8192x1024 f16) @ Wo(1024x64) -> f32 out
// one wave per 16x16 tile: 512 * 4 = 2048 waves
// ========================================================================
__global__ __launch_bounds__(256) void k_outproj(const _Float16* __restrict__ Ctx,
                                                 const _Float16* __restrict__ WoT,
                                                 float* __restrict__ out) {
  const int wid = blockIdx.x * (blockDim.x >> 5) + (threadIdx.x >> 5);
  const int nt = wid & 3;
  const int mt = wid >> 2;
  const int lane = threadIdx.x & 31;
  const int lc = lane & 15;
  const int hf = lane >> 4;

  v8f acc = {};
  const _Float16* arow = Ctx + (size_t)mt * 16 * HD;
  const _Float16* brow = WoT + (size_t)nt * 16 * HD;
#pragma unroll 4
  for (int k0 = 0; k0 < HD; k0 += 32) {
    v16h a = load_a16(arow + k0, HD);
    v16h b = load_b16(brow + k0, HD);
    acc = wmma_f16(a, b, acc);
  }
#pragma unroll
  for (int g = 0; g < 8; ++g) {
    int m = mt * 16 + g + 8 * hf;
    out[(size_t)m * DH + nt * 16 + lc] = acc[g];
  }
}

// ========================================================================
extern "C" void kernel_launch(void* const* d_in, const int* in_sizes, int n_in,
                              void* d_out, int out_size, void* d_ws, size_t ws_size,
                              hipStream_t stream) {
  (void)in_sizes; (void)n_in; (void)out_size; (void)ws_size;

  const float* X    = (const float*)d_in[0];
  // d_in[1] index, d_in[2] trait_weight: unused by forward
  const unsigned char* mask = (const unsigned char*)d_in[3];
  const float* Wq = (const float*)d_in[4];
  const float* bq = (const float*)d_in[5];
  const float* Wk = (const float*)d_in[6];
  const float* bk = (const float*)d_in[7];
  const float* Wv = (const float*)d_in[8];
  const float* bv = (const float*)d_in[9];
  const float* Wo = (const float*)d_in[10];
  float* out = (float*)d_out;

  // ---- workspace carve-up ----
  char* p = (char*)d_ws;
  auto carve = [&](size_t bytes) -> void* {
    void* r = (void*)p;
    p += (bytes + 255) & ~(size_t)255;
    return r;
  };
  _Float16* WqT = (_Float16*)carve((size_t)DD * HD * 2);   // 1 MB
  _Float16* WkT = (_Float16*)carve((size_t)DD * HD * 2);
  _Float16* WvT = (_Float16*)carve((size_t)DD * HD * 2);
  _Float16* WoT = (_Float16*)carve((size_t)HD * DH * 2);   // 128 KB
  _Float16* Qbf = (_Float16*)carve((size_t)MROWS * HD * 2); // 16 MB
  _Float16* Kbf = (_Float16*)carve((size_t)MROWS * HD * 2);
  _Float16* Vt  = (_Float16*)carve((size_t)MROWS * HD * 2);
  _Float16* Ctx = (_Float16*)carve((size_t)MROWS * HD * 2);

  // ---- stage 0: weight transpose + f16 convert ----
  {
    int total = DD * HD;  // 524288
    int blocks = (total + 255) / 256;
    k_transpose_w<<<blocks, 256, 0, stream>>>(Wq, WqT, DD, HD);
    k_transpose_w<<<blocks, 256, 0, stream>>>(Wk, WkT, DD, HD);
    k_transpose_w<<<blocks, 256, 0, stream>>>(Wv, WvT, DD, HD);
    int totalo = HD * DH; // 65536
    k_transpose_w<<<(totalo + 255) / 256, 256, 0, stream>>>(Wo, WoT, HD, DH);
  }

  // ---- stage 1: QKV projection (98304 waves / 8 per block) ----
  k_qkv<<<(MROWS / 16) * (HD / 16) * 3 / 8, 256, 0, stream>>>(
      X, WqT, WkT, WvT, bq, bk, bv, Qbf, Kbf, Vt);

  // ---- stage 2: flash attention (8192 waves) ----
  k_attn<<<BB * HH * (NN / 16) / 8, 256, 0, stream>>>(Qbf, Kbf, Vt, mask, Ctx);

  // ---- stage 3: output projection (2048 waves) ----
  k_outproj<<<(MROWS / 16) * (DH / 16) / 8, 256, 0, stream>>>(Ctx, WoT, out);
}
